// GPNStarColCrossAttention_5789615915683
// MI455X (gfx1250) — matrix operand order repaired and checked by reading, not compiled, not run on hardware
//
#include <hip/hip_runtime.h>

#define HIDDEN 768
#define NHEADS 6
#define HDIM   64
#define ALLH   384
#define BB     4
#define LL     256
#define SEQ    89
#define SP     96      // padded S (6 tiles of 16)

#define EBS 776        // embedding LDS row stride (halfs): 768 + 8 pad
#define QKS 72         // q/k LDS row stride (halfs): 64 + 8 pad
#define VTS 104        // vT LDS row stride (halfs): 96 + 8 pad
#define PSS 104        // probs LDS row stride (halfs)
#define SCS 104        // scores LDS row stride (floats)

// LDS byte offsets
#define EB_OFF 0
#define Q_OFF  148992          // 96*776*2
#define K_OFF  162816          // Q_OFF + 96*72*2
#define VT_OFF 176640          // K_OFF + 96*72*2
#define LDS_TOTAL 189952       // VT_OFF + 64*104*2
#define SC_OFF 0               // scores (f32) alias embedding buffer
#define P_OFF  40960           // probs (f16) alias embedding buffer (after scores)

typedef __attribute__((ext_vector_type(16))) _Float16 v16h;
typedef __attribute__((ext_vector_type(8)))  _Float16 v8h;
typedef __attribute__((ext_vector_type(4)))  _Float16 v4h;
typedef __attribute__((ext_vector_type(8)))  float    v8f;

union V16 { v16h v; v8h h[2]; };

// ---------------- kernel 1: weight convert + transpose ----------------
// W: [768][384] f32 row-major  ->  Wt: [384][768] f16 (row n holds column n of W)
__global__ __launch_bounds__(256) void cvt_w_kernel(const float* __restrict__ W,
                                                    _Float16* __restrict__ Wt) {
  const int n = blockIdx.x;
  for (int k = threadIdx.x; k < HIDDEN; k += 256)
    Wt[(size_t)n * HIDDEN + k] = (_Float16)W[(size_t)k * ALLH + n];
}

// A fragment (16x32 f16, M x K): lane lo = row m0+lo; hi selects K sub-block
__device__ __forceinline__ V16 loadA(const _Float16* base, int row, int stride,
                                     int k0, int hi) {
  V16 a;
  const _Float16* p = base + (size_t)row * stride + k0 + hi * 8;
  a.h[0] = *(const v8h*)p;
  a.h[1] = *(const v8h*)(p + 16);
  return a;
}

// B fragment (32x16 f16, K x N): rowptr = start of column-lane's K-major row
__device__ __forceinline__ V16 loadB(const _Float16* rowptr, int k0, int hi) {
  V16 b;
  const _Float16* p = rowptr + k0 + hi * 8;
  b.h[0] = *(const v8h*)p;
  b.h[1] = *(const v8h*)(p + 16);
  return b;
}

// ---------------- kernel 2: fused QKV projection + attention ----------------
__global__ __launch_bounds__(256) void gpn_attn_kernel(
    const float* __restrict__ hs, const float* __restrict__ src,
    const float* __restrict__ amask, const float* __restrict__ ebias,
    const _Float16* __restrict__ Wt,
    const float* __restrict__ bq, const float* __restrict__ bk,
    const float* __restrict__ bv, float* __restrict__ out) {
  extern __shared__ char smem[];
  _Float16* ebuf = (_Float16*)(smem + EB_OFF);  // 96 x 776 f16
  _Float16* qL   = (_Float16*)(smem + Q_OFF);   // 96 x 72  f16
  _Float16* kL   = (_Float16*)(smem + K_OFF);   // 96 x 72  f16
  _Float16* vT   = (_Float16*)(smem + VT_OFF);  // 64 x 104 f16 (transposed V)
  float*    sc   = (float*)(smem + SC_OFF);     // 96 x 104 f32 (aliases ebuf)
  _Float16* pL   = (_Float16*)(smem + P_OFF);   // 96 x 104 f16 (aliases ebuf)

  const int tid  = threadIdx.x;
  const int wave = tid >> 5;
  const int lane = tid & 31;
  const int lo   = lane & 15;
  const int hi   = lane >> 4;

  const int h  = blockIdx.x % NHEADS;
  const int bl = blockIdx.x / NHEADS;   // b*L + l ; heads consecutive -> L2 reuse
  const int hb = h * HDIM;

  const float* hsT  = hs  + (size_t)bl * SEQ * HIDDEN;
  const float* srcT = src + (size_t)bl * SEQ * HIDDEN;
  const float* amT  = amask + (size_t)bl * SEQ;
  const float* ebT  = ebias + (size_t)bl * SEQ;

  // zero pad rows 89..95 of the embedding buffer (once; never overwritten)
  {
    uint32_t* z = (uint32_t*)(ebuf + 89 * EBS);
    for (int i = tid; i < (7 * EBS) / 2; i += 256) z[i] = 0u;
  }

  auto fill_eb = [&](const float* g) {
    for (int i = tid; i < SEQ * (HIDDEN / 4); i += 256) {
      const int r = i / (HIDDEN / 4);
      const int c = i % (HIDDEN / 4);
      const float4 f = ((const float4*)g)[(size_t)r * (HIDDEN / 4) + c];
      v4h hv;
      hv.x = (_Float16)f.x; hv.y = (_Float16)f.y;
      hv.z = (_Float16)f.z; hv.w = (_Float16)f.w;
      *(v4h*)(ebuf + (size_t)r * EBS + c * 4) = hv;
    }
  };

  // 96x64 projection from ebuf (f16) x Wt-head-slice -> dst (f16)
  auto project = [&](const _Float16* Wmat, const float* bias, _Float16* dst,
                     int dstride, bool trans) {
    for (int t = wave * 3; t < wave * 3 + 3; ++t) {   // 24 tiles / 8 waves
      const int m0 = (t >> 2) * 16;
      const int n0 = (t & 3) * 16;
      v8f acc = {};
      const _Float16* wrow = Wmat + (size_t)(hb + n0 + lo) * HIDDEN;
#pragma unroll 4
      for (int k0 = 0; k0 < HIDDEN; k0 += 32) {
        V16 a = loadA(ebuf, m0 + lo, EBS, k0, hi);
        V16 b = loadB(wrow, k0, hi);
        acc = __builtin_amdgcn_wmma_f32_16x16x32_f16(false, a.v, false, b.v,
                                                     (short)0, acc, false, false);
      }
      const float bia = bias[hb + n0 + lo];
      if (!trans) {
        _Float16* d = dst + (size_t)(m0 + hi * 8) * dstride + n0 + lo;
#pragma unroll
        for (int j = 0; j < 8; ++j) d[(size_t)j * dstride] = (_Float16)(acc[j] + bia);
      } else {
        v8h hv;
#pragma unroll
        for (int j = 0; j < 8; ++j) hv[j] = (_Float16)(acc[j] + bia);
        *(v8h*)(dst + (size_t)(n0 + lo) * dstride + m0 + hi * 8) = hv;
      }
    }
  };

  // -------- phase 1: K and V from source embeddings --------
  fill_eb(srcT);
  __syncthreads();
  project(Wt + (size_t)1 * ALLH * HIDDEN, bk, kL, QKS, false);
  project(Wt + (size_t)2 * ALLH * HIDDEN, bv, vT, VTS, true);
  __syncthreads();

  // -------- phase 2: Q from hidden states --------
  fill_eb(hsT);
  __syncthreads();
  project(Wt, bq, qL, QKS, false);
  __syncthreads();

  // -------- phase 3: scores = Q K^T / 8 + bias --------
  for (int t = wave; t < 36; t += 8) {   // 6x6 tiles of 16x16
    const int m0 = (t / 6) * 16;
    const int n0 = (t % 6) * 16;
    v8f acc = {};
#pragma unroll
    for (int d0 = 0; d0 < HDIM; d0 += 32) {
      V16 a = loadA(qL, m0 + lo, QKS, d0, hi);
      V16 b = loadB(kL + (size_t)(n0 + lo) * QKS, d0, hi);  // row j of K = col j of K^T
      acc = __builtin_amdgcn_wmma_f32_16x16x32_f16(false, a.v, false, b.v,
                                                   (short)0, acc, false, false);
    }
    const int jc = n0 + lo;
    const float badd = (jc < SEQ) ? (amT[jc] + ebT[jc]) : -3e30f;
    float* srow = sc + (size_t)(m0 + hi * 8) * SCS + jc;
#pragma unroll
    for (int j = 0; j < 8; ++j) srow[(size_t)j * SCS] = acc[j] * 0.125f + badd;
  }
  __syncthreads();

  // -------- phase 4: softmax over k (rows of scores) --------
  if (tid < SP) {
    float* row = sc + (size_t)tid * SCS;
    float mx = -1e30f;
    for (int j = 0; j < SEQ; ++j) mx = fmaxf(mx, row[j]);
    float sum = 0.f;
    for (int j = 0; j < SEQ; ++j) {
      const float e = __expf(row[j] - mx);
      row[j] = e;
      sum += e;
    }
    const float inv = 1.0f / sum;
    _Float16* pr = pL + (size_t)tid * PSS;
    for (int j = 0; j < SEQ; ++j) pr[j] = (_Float16)(row[j] * inv);
    for (int j = SEQ; j < SP; ++j) pr[j] = (_Float16)0.f;
  }
  __syncthreads();

  // -------- phase 5: ctx = probs @ V --------
  for (int t = wave * 3; t < wave * 3 + 3; ++t) {   // 6x4 tiles of 16x16
    const int m0 = (t >> 2) * 16;
    const int n0 = (t & 3) * 16;
    v8f acc = {};
#pragma unroll
    for (int j0 = 0; j0 < SP; j0 += 32) {
      V16 a = loadA(pL, m0 + lo, PSS, j0, hi);
      V16 b = loadB(vT + (size_t)(n0 + lo) * VTS, j0, hi);  // vT row d = col d of V
      acc = __builtin_amdgcn_wmma_f32_16x16x32_f16(false, a.v, false, b.v,
                                                   (short)0, acc, false, false);
    }
    float* obase = out + (size_t)bl * SEQ * ALLH + hb + n0 + lo;
#pragma unroll
    for (int j = 0; j < 8; ++j) {
      const int m = m0 + hi * 8 + j;
      if (m < SEQ) obase[(size_t)m * ALLH] = acc[j];
    }
  }
}

extern "C" void kernel_launch(void* const* d_in, const int* in_sizes, int n_in,
                              void* d_out, int out_size, void* d_ws, size_t ws_size,
                              hipStream_t stream) {
  const float* hs  = (const float*)d_in[0];
  const float* src = (const float*)d_in[1];
  const float* am  = (const float*)d_in[2];
  const float* eb  = (const float*)d_in[3];
  const float* Wq  = (const float*)d_in[4];
  const float* bq  = (const float*)d_in[5];
  const float* Wk  = (const float*)d_in[6];
  const float* bk  = (const float*)d_in[7];
  const float* Wv  = (const float*)d_in[8];
  const float* bv  = (const float*)d_in[9];
  float* out = (float*)d_out;

  _Float16* Wt = (_Float16*)d_ws;  // [3][384][768] f16 = 1.77 MB

  cvt_w_kernel<<<ALLH, 256, 0, stream>>>(Wq, Wt);
  cvt_w_kernel<<<ALLH, 256, 0, stream>>>(Wk, Wt + (size_t)1 * ALLH * HIDDEN);
  cvt_w_kernel<<<ALLH, 256, 0, stream>>>(Wv, Wt + (size_t)2 * ALLH * HIDDEN);

  gpn_attn_kernel<<<BB * LL * NHEADS, 256, LDS_TOTAL, stream>>>(
      hs, src, am, eb, Wt, bq, bk, bv, out);
}
